// TransNet_48610439856825
// MI455X (gfx1250) — matrix operand order repaired and checked by reading, not compiled, not run on hardware
//
#include <hip/hip_runtime.h>

typedef unsigned short u16;
typedef unsigned int   u32;

typedef __attribute__((ext_vector_type(16))) __bf16 v16bf;
typedef __attribute__((ext_vector_type(8)))  float  v8f;

union BFrag { v16bf v; u16 u[16]; uint4 q[2]; };

__device__ inline u16 f2b(float f) {
  u32 x = __float_as_uint(f);
  u32 r = x + 0x7FFFu + ((x >> 16) & 1u);
  return (u16)(r >> 16);
}
__device__ inline float b2f(u16 h) { return __uint_as_float(((u32)h) << 16); }

// A fragment: 16x32 bf16 tile, row-major source [M][K] (lda in elements).
__device__ inline v16bf load_a16(const u16* __restrict__ A, int lda, int k0, int lane) {
  int m  = lane & 15;
  int kb = (lane & 16) ? 8 : 0;
  const u16* r = A + (size_t)m * lda + k0 + kb;
  BFrag f;
  f.q[0] = *(const uint4*)r;
  f.q[1] = *(const uint4*)(r + 16);
  return f.v;
}

// B fragment from TRANSPOSED source BT[Ncols][Kd] (ldbt in elements).
__device__ inline v16bf load_bT16(const u16* __restrict__ BT, int ldbt, int n0, int k0, int lane) {
  int col = n0 + (lane & 15);
  int kb  = (lane & 16) ? 8 : 0;
  const u16* p = BT + (size_t)col * ldbt + k0 + kb;
  BFrag f;
  f.q[0] = *(const uint4*)p;
  f.q[1] = *(const uint4*)(p + 16);
  return f.v;
}

__device__ inline v8f wmma_bf16(v16bf a, v16bf b, v8f c) {
  return __builtin_amdgcn_wmma_f32_16x16x32_bf16(false, a, false, b, (short)0, c, false, false);
}

__device__ inline v8f v8zero() { v8f r = {0.f,0.f,0.f,0.f,0.f,0.f,0.f,0.f}; return r; }

__device__ inline float act_apply(float v, int act) {
  if (act == 1) return v > 0.f ? v : 0.f;
  if (act == 2) return v > 0.f ? v : 0.2f * v;
  return v;
}

// ---- CDNA5 async global->LDS copy (b128 per lane, tracked by ASYNCcnt) ----
// LDS dest address = low 32 bits of the flat shared pointer (aperture rule:
// flat LDS addresses are workgroup-relative in addr[31:0]).
__device__ inline void async_copy_b128(u32 lds_off, const void* gaddr) {
  asm volatile("global_load_async_to_lds_b128 %0, %1, off"
               :: "v"(lds_off), "v"(gaddr) : "memory");
}
__device__ inline void wait_async0() {
  asm volatile("s_wait_asynccnt 0" ::: "memory");
}

// ---------------------------------------------------------------------------
// Conversion kernels
// ---------------------------------------------------------------------------
__global__ void cvt_kernel(const float* __restrict__ src, u16* __restrict__ dst) {
  size_t t = (size_t)blockIdx.x * blockDim.x + threadIdx.x;  // grid covers exactly
  dst[t] = f2b(src[t]);
}

// W[b][rows][cols] f32 -> WT[b][cols][rows] bf16
__global__ void cvt_t_kernel(const float* __restrict__ W, u16* __restrict__ WT,
                             int rows, int cols) {
  int b = blockIdx.y;
  size_t t = (size_t)blockIdx.x * blockDim.x + threadIdx.x;
  size_t nel = (size_t)rows * cols;
  if (t >= nel) return;
  int r = (int)(t / cols), c = (int)(t % cols);
  WT[(size_t)b * nel + (size_t)c * rows + r] = f2b(W[(size_t)b * nel + t]);
}

// ---------------------------------------------------------------------------
// Generic WMMA GEMM: C tile = 64x64 per block (4 waves, 4 col tiles each).
// A bf16 row-major [M][Kd], B bf16 TRANSPOSED [64][Kd]. N fixed = 64.
// MODE 0: bf16 row-major store (+bias/act)
// MODE 1: dimRedu: rows ordered (n*64+s) -> f32 lin[s][n][c] + bf16 copy
// MODE 2: bf16 store transposed within 1024-row batches: CT[s][col][k]
// ---------------------------------------------------------------------------
template <int MODE>
__global__ __launch_bounds__(128) void gemm64(
    const u16* __restrict__ A, long strideA, int lda,
    const u16* __restrict__ BT, long strideB, int ldbt,
    const float* __restrict__ bias, int act,
    u16* __restrict__ Cb, long strideC,
    float* __restrict__ Cf, u16* __restrict__ Cb2, int Kd)
{
  int lane = threadIdx.x & 31, w = threadIdx.x >> 5;
  int mblk = blockIdx.x, batch = blockIdx.y;
  const u16* Ab = A + (size_t)batch * strideA + ((size_t)mblk * 64 + w * 16) * lda;
  const u16* Bb = BT + (size_t)batch * strideB;
  v8f c0 = v8zero(), c1 = v8zero(), c2 = v8zero(), c3 = v8zero();
  for (int k0 = 0; k0 < Kd; k0 += 32) {
    v16bf a = load_a16(Ab, lda, k0, lane);
    c0 = wmma_bf16(a, load_bT16(Bb, ldbt,  0, k0, lane), c0);
    c1 = wmma_bf16(a, load_bT16(Bb, ldbt, 16, k0, lane), c1);
    c2 = wmma_bf16(a, load_bT16(Bb, ldbt, 32, k0, lane), c2);
    c3 = wmma_bf16(a, load_bT16(Bb, ldbt, 48, k0, lane), c3);
  }
  int rowb = mblk * 64 + w * 16 + ((lane & 16) ? 8 : 0);
  int col0 = lane & 15;
  v8f cc[4] = {c0, c1, c2, c3};
#pragma unroll
  for (int nt = 0; nt < 4; ++nt) {
#pragma unroll
    for (int i = 0; i < 8; ++i) {
      int row = rowb + i;
      int col = nt * 16 + col0;
      float v = cc[nt][i];
      if (bias) v += bias[col];
      v = act_apply(v, act);
      if (MODE == 0) {
        Cb[(size_t)batch * strideC + (size_t)row * 64 + col] = f2b(v);
      } else if (MODE == 1) {
        int n = row >> 6, s = row & 63;
        size_t id = ((size_t)s * 8192 + n) * 64 + col;
        Cf[id] = v;
        Cb2[id] = f2b(v);
      } else {  // MODE 2: transposed within s-batches of 1024 rows
        int s = row >> 10, k = row & 1023;
        Cb[(size_t)s * 65536 + (size_t)col * 1024 + k] = f2b(v);
      }
    }
  }
}

// ---------------------------------------------------------------------------
// Batched adj GEMM with async global->LDS panel staging (CDNA5 path).
// C[batch][1024][64] = adj[batch][1024][1024] @ B, B given transposed [64][1024].
// Per block: 64x64 C tile; K loop in 64-wide panels staged into LDS with
// GLOBAL_LOAD_ASYNC_TO_LDS_B128, drained by s_wait_asynccnt.
// ---------------------------------------------------------------------------
__global__ __launch_bounds__(128) void gemm_adj_async(
    const u16* __restrict__ A, const u16* __restrict__ BT, u16* __restrict__ C)
{
  __shared__ __align__(16) u16 sa[64 * 64];   // 8KB A panel
  __shared__ __align__(16) u16 sb[64 * 64];   // 8KB BT panel
  int tid = threadIdx.x, lane = tid & 31, w = tid >> 5;
  int mblk = blockIdx.x, batch = blockIdx.y;
  const u16* Ab = A + (size_t)batch * 1048576 + (size_t)mblk * 64 * 1024;
  const u16* Bb = BT + (size_t)batch * 65536;
  u32 sa_base = (u32)(size_t)sa;
  u32 sb_base = (u32)(size_t)sb;

  v8f c0 = v8zero(), c1 = v8zero(), c2 = v8zero(), c3 = v8zero();
  for (int kk = 0; kk < 1024; kk += 64) {
    // stage 8KB + 8KB panels: 512 chunks of 16B each per panel, 4 per thread
#pragma unroll
    for (int i = 0; i < 4; ++i) {
      int chunk = tid + 128 * i;        // 0..511
      int row = chunk >> 3;             // panel row (A row / BT col)
      int ko  = (chunk & 7) * 8;        // element offset within 64-wide panel
      u32 loff = (u32)((row * 64 + ko) * 2);
      async_copy_b128(sa_base + loff, Ab + (size_t)row * 1024 + kk + ko);
      async_copy_b128(sb_base + loff, Bb + (size_t)row * 1024 + kk + ko);
    }
    wait_async0();
    __syncthreads();
#pragma unroll
    for (int k0 = 0; k0 < 64; k0 += 32) {
      v16bf a = load_a16(sa + w * 16 * 64, 64, k0, lane);
      c0 = wmma_bf16(a, load_bT16(sb, 64,  0, k0, lane), c0);
      c1 = wmma_bf16(a, load_bT16(sb, 64, 16, k0, lane), c1);
      c2 = wmma_bf16(a, load_bT16(sb, 64, 32, k0, lane), c2);
      c3 = wmma_bf16(a, load_bT16(sb, 64, 48, k0, lane), c3);
    }
    __syncthreads();
  }

  int rowb = mblk * 64 + w * 16 + ((lane & 16) ? 8 : 0);
  int col0 = lane & 15;
  v8f cc[4] = {c0, c1, c2, c3};
#pragma unroll
  for (int nt = 0; nt < 4; ++nt)
#pragma unroll
    for (int i = 0; i < 8; ++i)
      C[(size_t)batch * 65536 + (size_t)(rowb + i) * 64 + nt * 16 + col0] = f2b(cc[nt][i]);
}

// ---------------------------------------------------------------------------
// Gather / scatter / zero helpers
// ---------------------------------------------------------------------------
__global__ void gather_kernel(const u16* __restrict__ linbf, const int* __restrict__ idx,
                              u16* __restrict__ gT) {
  size_t t = (size_t)blockIdx.x * blockDim.x + threadIdx.x;   // S*K*HD threads exactly
  int c = (int)(t & 63);
  size_t rk = t >> 6;
  int k = (int)(rk & 1023);
  int s = (int)(rk >> 10);
  int node = idx[s * 1024 + k];
  gT[(size_t)s * 65536 + (size_t)c * 1024 + k] = linbf[((size_t)s * 8192 + node) * 64 + c];
}

__global__ void scatter_kernel(const u16* __restrict__ r, const int* __restrict__ idx,
                               u16* __restrict__ z) {
  size_t t = (size_t)blockIdx.x * blockDim.x + threadIdx.x;   // S*K*TD threads exactly
  int c = (int)(t & 63);
  size_t rk = t >> 6;
  int k = (int)(rk & 1023);
  int s = (int)(rk >> 10);
  int node = idx[s * 1024 + k];
  z[((size_t)node * 64 + s) * 64 + c] = r[t];
}

__global__ void zero_kernel(u32* __restrict__ p) {
  size_t t = (size_t)blockIdx.x * blockDim.x + threadIdx.x;
  p[t] = 0u;
}

// ---------------------------------------------------------------------------
// Fused per-node transformer. All weight operands are pre-transposed bf16.
// ---------------------------------------------------------------------------
__device__ inline void ln_row(u16* h, int row, const float* g, const float* b) {
  float m = 0.f;
  for (int j = 0; j < 128; ++j) m += b2f(h[row * 128 + j]);
  m *= (1.f / 128.f);
  float var = 0.f;
  for (int j = 0; j < 128; ++j) { float d = b2f(h[row * 128 + j]) - m; var += d * d; }
  var *= (1.f / 128.f);
  float rs = rsqrtf(var + 1e-5f);
  for (int j = 0; j < 128; ++j) {
    float v = (b2f(h[row * 128 + j]) - m) * rs * g[j] + b[j];
    h[row * 128 + j] = f2b(v);
  }
}

__global__ __launch_bounds__(128) void transformer_kernel(
    const u16* __restrict__ z,
    const u16* __restrict__ tinT, const float* __restrict__ tin_b,
    const u16* __restrict__ wqT, const float* __restrict__ bq,
    const u16* __restrict__ wkT, const float* __restrict__ bk,
    const u16* __restrict__ wvT, const float* __restrict__ bv,
    const u16* __restrict__ woT, const float* __restrict__ bo,
    const float* __restrict__ ln1_g, const float* __restrict__ ln1_b,
    const u16* __restrict__ f1T, const float* __restrict__ ff_b1,
    const u16* __restrict__ f2T, const float* __restrict__ ff_b2,
    const float* __restrict__ ln2_g, const float* __restrict__ ln2_b,
    float* __restrict__ rep)
{
  __shared__ __align__(16) u16 sh_h[64 * 128];   // 16KB hidden (bf16)
  __shared__ __align__(16) u16 sh_o[64 * 128];   // 16KB attn output (bf16)
  __shared__ __align__(16) char arena[32768];    // 32KB phase-overlaid scratch
  u16*   qh  = (u16*)arena;                      // [64][32]  q (A of scores)
  u16*   kh  = (u16*)(arena + 4096);             // [64][32]  k (== BT of scores)
  u16*   vhT = (u16*)(arena + 8192);             // [32][64]  v transposed (BT of o_h)
  float* sc  = (float*)(arena + 12288);          // [64][64]  scores f32
  u16*   pb  = (u16*)arena;                      // [64][64]  probs (reuses qh/kh)
  u16*   ffi = (u16*)arena;                      // [64][256] FF intermediate

  int tid  = threadIdx.x;
  int lane = tid & 31, w = tid >> 5;
  int rowb = w * 16 + ((lane & 16) ? 8 : 0);
  int col0 = lane & 15;
  size_t n = blockIdx.x;
  const u16* zn = z + n * 64 * 64;

  // ---- input projection: h = z_n @ tin_w + tin_b ----
  {
    v8f c[8];
#pragma unroll
    for (int i = 0; i < 8; ++i) c[i] = v8zero();
    for (int k0 = 0; k0 < 64; k0 += 32) {
      v16bf a = load_a16(zn + (size_t)w * 16 * 64, 64, k0, lane);
#pragma unroll
      for (int nt = 0; nt < 8; ++nt)
        c[nt] = wmma_bf16(a, load_bT16(tinT, 64, nt * 16, k0, lane), c[nt]);
    }
#pragma unroll
    for (int nt = 0; nt < 8; ++nt)
#pragma unroll
      for (int i = 0; i < 8; ++i) {
        int col = nt * 16 + col0;
        sh_h[(rowb + i) * 128 + col] = f2b(c[nt][i] + tin_b[col]);
      }
  }
  __syncthreads();

  for (int l = 0; l < 2; ++l) {
    const u16* wql = wqT + (size_t)l * 128 * 128;
    const u16* wkl = wkT + (size_t)l * 128 * 128;
    const u16* wvl = wvT + (size_t)l * 128 * 128;
    const u16* wol = woT + (size_t)l * 128 * 128;
    const u16* f1l = f1T + (size_t)l * 128 * 256;
    const u16* f2l = f2T + (size_t)l * 256 * 128;

    for (int hh = 0; hh < 4; ++hh) {
      // q_h -> qh [64][32]
      {
        v8f c[2] = {v8zero(), v8zero()};
        for (int k0 = 0; k0 < 128; k0 += 32) {
          v16bf a = load_a16(sh_h + w * 16 * 128, 128, k0, lane);
#pragma unroll
          for (int nt = 0; nt < 2; ++nt)
            c[nt] = wmma_bf16(a, load_bT16(wql, 128, hh * 32 + nt * 16, k0, lane), c[nt]);
        }
#pragma unroll
        for (int nt = 0; nt < 2; ++nt)
#pragma unroll
          for (int i = 0; i < 8; ++i) {
            int cl = nt * 16 + col0;
            qh[(rowb + i) * 32 + cl] = f2b(c[nt][i] + bq[l * 128 + hh * 32 + cl]);
          }
      }
      // k_h -> kh [64][32] (plain; IS the transposed-B layout for scores)
      {
        v8f c[2] = {v8zero(), v8zero()};
        for (int k0 = 0; k0 < 128; k0 += 32) {
          v16bf a = load_a16(sh_h + w * 16 * 128, 128, k0, lane);
#pragma unroll
          for (int nt = 0; nt < 2; ++nt)
            c[nt] = wmma_bf16(a, load_bT16(wkl, 128, hh * 32 + nt * 16, k0, lane), c[nt]);
        }
#pragma unroll
        for (int nt = 0; nt < 2; ++nt)
#pragma unroll
          for (int i = 0; i < 8; ++i) {
            int cl = nt * 16 + col0;
            kh[(rowb + i) * 32 + cl] = f2b(c[nt][i] + bk[l * 128 + hh * 32 + cl]);
          }
      }
      // v_h -> vhT [32][64] (transposed; BT layout for o_h = probs @ v)
      {
        v8f c[2] = {v8zero(), v8zero()};
        for (int k0 = 0; k0 < 128; k0 += 32) {
          v16bf a = load_a16(sh_h + w * 16 * 128, 128, k0, lane);
#pragma unroll
          for (int nt = 0; nt < 2; ++nt)
            c[nt] = wmma_bf16(a, load_bT16(wvl, 128, hh * 32 + nt * 16, k0, lane), c[nt]);
        }
#pragma unroll
        for (int nt = 0; nt < 2; ++nt)
#pragma unroll
          for (int i = 0; i < 8; ++i) {
            int cl = nt * 16 + col0;
            vhT[cl * 64 + (rowb + i)] = f2b(c[nt][i] + bv[l * 128 + hh * 32 + cl]);
          }
      }
      __syncthreads();

      // scores = (q_h @ k_h^T) / sqrt(dh)
      {
        v8f c[4] = {v8zero(), v8zero(), v8zero(), v8zero()};
        v16bf a = load_a16(qh + w * 16 * 32, 32, 0, lane);
#pragma unroll
        for (int nt = 0; nt < 4; ++nt)
          c[nt] = wmma_bf16(a, load_bT16(kh, 32, nt * 16, 0, lane), c[nt]);
#pragma unroll
        for (int nt = 0; nt < 4; ++nt)
#pragma unroll
          for (int i = 0; i < 8; ++i)
            sc[(rowb + i) * 64 + nt * 16 + col0] = c[nt][i] * 0.17677669529663687f;
      }
      __syncthreads();

      // row softmax -> pb bf16
      if (tid < 64) {
        float m = -1e30f;
        for (int j = 0; j < 64; ++j) m = fmaxf(m, sc[tid * 64 + j]);
        float ssum = 0.f;
        for (int j = 0; j < 64; ++j) ssum += __expf(sc[tid * 64 + j] - m);
        float inv = 1.f / ssum;
        for (int j = 0; j < 64; ++j) pb[tid * 64 + j] = f2b(__expf(sc[tid * 64 + j] - m) * inv);
      }
      __syncthreads();

      // o_h = probs @ v_h -> sh_o columns [hh*32, hh*32+32)
      {
        v8f c[2] = {v8zero(), v8zero()};
        for (int k0 = 0; k0 < 64; k0 += 32) {
          v16bf a = load_a16(pb + w * 16 * 64, 64, k0, lane);
#pragma unroll
          for (int nt = 0; nt < 2; ++nt)
            c[nt] = wmma_bf16(a, load_bT16(vhT, 64, nt * 16, k0, lane), c[nt]);
        }
#pragma unroll
        for (int nt = 0; nt < 2; ++nt)
#pragma unroll
          for (int i = 0; i < 8; ++i)
            sh_o[(rowb + i) * 128 + hh * 32 + nt * 16 + col0] = f2b(c[nt][i]);
      }
      __syncthreads();
    }

    // h = h + o @ wo + bo  (in-place)
    {
      v8f c[8];
#pragma unroll
      for (int i = 0; i < 8; ++i) c[i] = v8zero();
      for (int k0 = 0; k0 < 128; k0 += 32) {
        v16bf a = load_a16(sh_o + w * 16 * 128, 128, k0, lane);
#pragma unroll
        for (int nt = 0; nt < 8; ++nt)
          c[nt] = wmma_bf16(a, load_bT16(wol, 128, nt * 16, k0, lane), c[nt]);
      }
#pragma unroll
      for (int nt = 0; nt < 8; ++nt)
#pragma unroll
        for (int i = 0; i < 8; ++i) {
          int col = nt * 16 + col0;
          int id  = (rowb + i) * 128 + col;
          float v = c[nt][i] + bo[l * 128 + col] + b2f(sh_h[id]);
          sh_h[id] = f2b(v);
        }
    }
    __syncthreads();
    if (tid < 64) ln_row(sh_h, tid, ln1_g + l * 128, ln1_b + l * 128);
    __syncthreads();

    // ffi = relu(h @ ff_w1 + b1)  [64][256]
    for (int t = 0; t < 16; ++t) {
      v8f c = v8zero();
      for (int k0 = 0; k0 < 128; k0 += 32) {
        v16bf a = load_a16(sh_h + w * 16 * 128, 128, k0, lane);
        c = wmma_bf16(a, load_bT16(f1l, 128, t * 16, k0, lane), c);
      }
#pragma unroll
      for (int i = 0; i < 8; ++i) {
        int col = t * 16 + col0;
        float v = c[i] + ff_b1[l * 256 + col];
        ffi[(rowb + i) * 256 + col] = f2b(v > 0.f ? v : 0.f);
      }
    }
    __syncthreads();

    // h = h + ffi @ ff_w2 + b2 (in-place)
    {
      v8f c[8];
#pragma unroll
      for (int i = 0; i < 8; ++i) c[i] = v8zero();
      for (int k0 = 0; k0 < 256; k0 += 32) {
        v16bf a = load_a16(ffi + w * 16 * 256, 256, k0, lane);
#pragma unroll
        for (int nt = 0; nt < 8; ++nt)
          c[nt] = wmma_bf16(a, load_bT16(f2l, 256, nt * 16, k0, lane), c[nt]);
      }
#pragma unroll
      for (int nt = 0; nt < 8; ++nt)
#pragma unroll
        for (int i = 0; i < 8; ++i) {
          int col = nt * 16 + col0;
          int id  = (rowb + i) * 128 + col;
          float v = c[nt][i] + ff_b2[l * 128 + col] + b2f(sh_h[id]);
          sh_h[id] = f2b(v);
        }
    }
    __syncthreads();
    if (tid < 64) ln_row(sh_h, tid, ln2_g + l * 128, ln2_b + l * 128);
    __syncthreads();
  }

  // rep[n] = mean over sequence
  if (tid < 128) {
    float sum = 0.f;
    for (int r = 0; r < 64; ++r) sum += b2f(sh_h[r * 128 + tid]);
    rep[n * 128 + tid] = sum * (1.f / 64.f);
  }
}

// ---------------------------------------------------------------------------
// Small heads (classifier + domain discriminators)
// ---------------------------------------------------------------------------
__device__ inline void lsm2(float z0, float z1, float* o) {
  float m = fmaxf(z0, z1);
  float lse = m + __logf(__expf(z0 - m) + __expf(z1 - m));
  o[0] = z0 - lse;
  o[1] = z1 - lse;
}

__global__ void head_kernel(const float* __restrict__ rep,
                            const float* __restrict__ out_w, const float* __restrict__ out_b,
                            const float* __restrict__ dd_w1, const float* __restrict__ dd_b1,
                            const float* __restrict__ dd_w2, const float* __restrict__ dd_b2,
                            float* __restrict__ out_cls, float* __restrict__ out_dom)
{
  int n = blockIdx.x * blockDim.x + threadIdx.x;   // exactly 8192 threads
  const float* r = rep + (size_t)n * 128;
  float z0 = out_b[0], z1 = out_b[1];
  for (int c = 0; c < 128; ++c) {
    z0 += r[c] * out_w[c * 2 + 0];
    z1 += r[c] * out_w[c * 2 + 1];
  }
  lsm2(z0, z1, out_cls + (size_t)n * 2);

  float d0 = dd_b2[0], d1 = dd_b2[1];
  for (int j = 0; j < 128; ++j) {
    float h = dd_b1[j];
    for (int c = 0; c < 128; ++c) h += r[c] * dd_w1[c * 128 + j];
    h = h > 0.f ? h : 0.2f * h;
    d0 += h * dd_w2[j * 2 + 0];
    d1 += h * dd_w2[j * 2 + 1];
  }
  lsm2(d0, d1, out_dom + (size_t)n * 2);
}

__global__ void ddl2_kernel(const u16* __restrict__ hid,
                            const float* __restrict__ w2, const float* __restrict__ b2,
                            float* __restrict__ out)
{
  size_t i = (size_t)blockIdx.x * blockDim.x + threadIdx.x;  // exactly S*N threads
  const u16* h = hid + i * 64;
  float z0 = b2[0], z1 = b2[1];
  for (int j = 0; j < 64; ++j) {
    float v = b2f(h[j]);
    z0 += v * w2[j * 2 + 0];
    z1 += v * w2[j * 2 + 1];
  }
  lsm2(z0, z1, out + i * 2);
}

// ---------------------------------------------------------------------------
// Launch
// ---------------------------------------------------------------------------
extern "C" void kernel_launch(void* const* d_in, const int* in_sizes, int n_in,
                              void* d_out, int out_size, void* d_ws, size_t ws_size,
                              hipStream_t stream) {
  const float* s_x   = (const float*)d_in[0];
  const float* t_x   = (const float*)d_in[1];
  const float* s_adj = (const float*)d_in[2];
  const float* t_adj = (const float*)d_in[3];
  const int*   s_idx = (const int*)d_in[4];
  const int*   t_idx = (const int*)d_in[5];
  const float* dr_ws = (const float*)d_in[7];
  const float* dr_bs = (const float*)d_in[8];
  const float* dr_wt = (const float*)d_in[9];
  const float* dr_bt = (const float*)d_in[10];
  const float* gnn_w1 = (const float*)d_in[11];
  const float* gnn_b1 = (const float*)d_in[12];
  const float* gnn_w2 = (const float*)d_in[13];
  const float* gnn_b2 = (const float*)d_in[14];
  const float* tin_w = (const float*)d_in[15];
  const float* tin_b = (const float*)d_in[16];
  const float* wq = (const float*)d_in[17];
  const float* bq = (const float*)d_in[18];
  const float* wk = (const float*)d_in[19];
  const float* bk = (const float*)d_in[20];
  const float* wv = (const float*)d_in[21];
  const float* bv = (const float*)d_in[22];
  const float* wo = (const float*)d_in[23];
  const float* bo = (const float*)d_in[24];
  const float* ln1_g = (const float*)d_in[25];
  const float* ln1_b = (const float*)d_in[26];
  const float* ff_w1 = (const float*)d_in[27];
  const float* ff_b1 = (const float*)d_in[28];
  const float* ff_w2 = (const float*)d_in[29];
  const float* ff_b2 = (const float*)d_in[30];
  const float* ln2_g = (const float*)d_in[31];
  const float* ln2_b = (const float*)d_in[32];
  const float* out_w = (const float*)d_in[33];
  const float* out_b = (const float*)d_in[34];
  const float* dd_w1 = (const float*)d_in[35];
  const float* dd_b1 = (const float*)d_in[36];
  const float* dd_w2 = (const float*)d_in[37];
  const float* dd_b2 = (const float*)d_in[38];
  const float* ddl_w1 = (const float*)d_in[39];
  const float* ddl_b1 = (const float*)d_in[40];
  const float* ddl_w2 = (const float*)d_in[41];
  const float* ddl_b2 = (const float*)d_in[42];

  float* out = (float*)d_out;
  float* rep_s  = out;
  float* rep_t  = out + 1048576;
  float* lin_s  = out + 2097152;
  float* lin_t  = out + 35651584;
  float* sdom   = out + 69206016;
  float* tdom   = out + 69222400;
  float* sdoml  = out + 69238784;
  float* tdoml  = out + 70287360;
  float* out_s  = out + 71335936;
  float* out_t  = out + 71352320;

  char* ws = (char*)d_ws;
  u16* stage = (u16*)(ws);                 // x_bf / adj_bf staging     134 MB
  u16* gT    = (u16*)(ws + 134217728);     // gathered, B-transposed      8 MB
  u16* ag    = (u16*)(ws + 142606336);     // adj@g                       8 MB
  u16* h1T   = (u16*)(ws + 150994944);     // relu(..) transposed         8 MB
  u16* ah1   = (u16*)(ws + 159383552);     // adj@h1                      8 MB
  u16* rbf   = (u16*)(ws + 167772160);     // GCN output                  8 MB
  u16* zbf   = (u16*)(ws + 176160768);     // [N,S,TD] bf16              64 MB
  u16* hid   = (u16*)(ws + 243269632);     // ddl hidden                 64 MB
  u16* linbf = (u16*)(ws + 310378496);     // bf16 copy of lin           64 MB
  char* wb   = ws + 377487360;             // transposed bf16 weights  ~600 KB
  u16* drsT = (u16*)(wb +      0);   // [64][128]
  u16* drtT = (u16*)(wb +  16384);   // [64][128]
  u16* g1T  = (u16*)(wb +  32768);   // [64][64]
  u16* g2T  = (u16*)(wb +  40960);   // [64][64]
  u16* tinT = (u16*)(wb +  49152);   // [128][64]
  u16* wqT  = (u16*)(wb +  65536);   // [2][128][128]
  u16* wkT  = (u16*)(wb + 131072);
  u16* wvT  = (u16*)(wb + 196608);
  u16* woT  = (u16*)(wb + 262144);
  u16* f1T  = (u16*)(wb + 327680);   // [2][256][128]
  u16* f2T  = (u16*)(wb + 458752);   // [2][128][256]
  u16* ddlT = (u16*)(wb + 589824);   // [64][64]

  // ---- one-time weight transpose+convert (tiny, L2-resident afterwards) ----
  cvt_t_kernel<<<dim3(32, 1), 256, 0, stream>>>(dr_ws, drsT, 128, 64);
  cvt_t_kernel<<<dim3(32, 1), 256, 0, stream>>>(dr_wt, drtT, 128, 64);
  cvt_t_kernel<<<dim3(16, 1), 256, 0, stream>>>(gnn_w1, g1T, 64, 64);
  cvt_t_kernel<<<dim3(16, 1), 256, 0, stream>>>(gnn_w2, g2T, 64, 64);
  cvt_t_kernel<<<dim3(32, 1), 256, 0, stream>>>(tin_w, tinT, 64, 128);
  cvt_t_kernel<<<dim3(64, 2), 256, 0, stream>>>(wq, wqT, 128, 128);
  cvt_t_kernel<<<dim3(64, 2), 256, 0, stream>>>(wk, wkT, 128, 128);
  cvt_t_kernel<<<dim3(64, 2), 256, 0, stream>>>(wv, wvT, 128, 128);
  cvt_t_kernel<<<dim3(64, 2), 256, 0, stream>>>(wo, woT, 128, 128);
  cvt_t_kernel<<<dim3(128, 2), 256, 0, stream>>>(ff_w1, f1T, 128, 256);
  cvt_t_kernel<<<dim3(128, 2), 256, 0, stream>>>(ff_w2, f2T, 256, 128);
  cvt_t_kernel<<<dim3(16, 1), 256, 0, stream>>>(ddl_w1, ddlT, 64, 64);

  auto run_branch = [&](const float* x, const float* adj, const int* idx,
                        const u16* drT, const float* drb,
                        float* lin, float* rep, float* dom, float* doml, float* outc) {
    // 1) x -> bf16 staging, then dimRedu GEMM with fused transpose store
    cvt_kernel<<<262144, 256, 0, stream>>>(x, stage);
    gemm64<1><<<dim3(8192, 1), 128, 0, stream>>>(
        stage, 0L, 128, drT, 0L, 128, drb, 0, nullptr, 0L, lin, linbf, 128);
    // 2) gather active nodes into B-transposed layout
    gather_kernel<<<16384, 256, 0, stream>>>(linbf, idx, gT);
    // 3) adj -> bf16 staging (x no longer needed)
    cvt_kernel<<<262144, 256, 0, stream>>>(adj, stage);
    // 4) ag = adj @ g   (async-LDS staged, batched over S)
    gemm_adj_async<<<dim3(16, 64), 128, 0, stream>>>(stage, gT, ag);
    // 5) h1 = relu(ag @ gnn_w1 + b1), stored transposed per s-batch
    gemm64<2><<<dim3(1024, 1), 128, 0, stream>>>(
        ag, 0L, 64, g1T, 0L, 64, gnn_b1, 1, h1T, 0L, nullptr, nullptr, 64);
    // 6) ah1 = adj @ h1
    gemm_adj_async<<<dim3(16, 64), 128, 0, stream>>>(stage, h1T, ah1);
    // 7) r = relu(ah1 @ gnn_w2 + b2)
    gemm64<0><<<dim3(1024, 1), 128, 0, stream>>>(
        ah1, 0L, 64, g2T, 0L, 64, gnn_b2, 1, rbf, 0L, nullptr, nullptr, 64);
    // 8) zero z, scatter r -> z[n][s][:]
    zero_kernel<<<65536, 256, 0, stream>>>((u32*)zbf);
    scatter_kernel<<<16384, 256, 0, stream>>>(rbf, idx, zbf);
    // 9) fused per-node transformer -> rep
    transformer_kernel<<<8192, 128, 0, stream>>>(
        zbf, tinT, tin_b, wqT, bq, wkT, bk, wvT, bv, woT, bo,
        ln1_g, ln1_b, f1T, ff_b1, f2T, ff_b2, ln2_g, ln2_b, rep);
    // 10) classifier + node-level domain discriminator
    head_kernel<<<32, 256, 0, stream>>>(rep, out_w, out_b, dd_w1, dd_b1, dd_w2, dd_b2,
                                        outc, dom);
    // 11) ddl layer 1 (leaky relu) via WMMA, layer 2 + log_softmax
    gemm64<0><<<dim3(8192, 1), 128, 0, stream>>>(
        linbf, 0L, 64, ddlT, 0L, 64, ddl_b1, 2, hid, 0L, nullptr, nullptr, 64);
    ddl2_kernel<<<2048, 256, 0, stream>>>(hid, ddl_w2, ddl_b2, doml);
  };

  run_branch(s_x, s_adj, s_idx, drsT, dr_bs, lin_s, rep_s, sdom, sdoml, out_s);
  run_branch(t_x, t_adj, t_idx, drtT, dr_bt, lin_t, rep_t, tdom, tdoml, out_t);
}
